// TopKDispatchMoE_73315091743521
// MI455X (gfx1250) — matrix operand order repaired and checked by reading, not compiled, not run on hardware
//
#include <hip/hip_runtime.h>
#include <math.h>

// Problem constants (match reference)
#define N_TOK 8192
#define HDIM  1024
#define NEXP  8
#define H2    (2 * HDIM)
#define TM    64                 // token tile: four 16-row WMMA row-groups
#define BLOCK 512                // 16 column-waves; each wave does all 4 row-groups
#define HCHUNK 512               // fc1 column chunk held in LDS
#define XS_STRIDE (HDIM + 8)     // bf16 elems, padded: conflict-free ds_load_b128
#define HS_STRIDE (HCHUNK + 8)
#define MAXT (N_TOK / TM)        // 128 tiles per expert upper bound

typedef __attribute__((ext_vector_type(16))) __bf16 bf16x16;
typedef __attribute__((ext_vector_type(8)))  __bf16 bf16x8;
typedef __attribute__((ext_vector_type(8)))  float  f32x8;

// ---------------------------------------------------------------------------
// Kernel 0: zero output + per-expert counters
// ---------------------------------------------------------------------------
__global__ void moe_zero(float* __restrict__ out, int* __restrict__ counts, int total) {
    int i = blockIdx.x * blockDim.x + threadIdx.x;
    if (i < total) out[i] = 0.0f;
    if (i < NEXP)  counts[i] = 0;
}

// ---------------------------------------------------------------------------
// Kernel 0b: fp32 -> bf16 weight conversion (8 elems / thread)
// ---------------------------------------------------------------------------
__global__ void moe_cvt_bf16(const float* __restrict__ src, __bf16* __restrict__ dst, int n) {
    int i = (blockIdx.x * blockDim.x + threadIdx.x) * 8;
    if (i + 8 <= n) {
        float4 a = *(const float4*)&src[i];
        float4 b = *(const float4*)&src[i + 4];
        bf16x8 v;
        v[0]=(__bf16)a.x; v[1]=(__bf16)a.y; v[2]=(__bf16)a.z; v[3]=(__bf16)a.w;
        v[4]=(__bf16)b.x; v[5]=(__bf16)b.y; v[6]=(__bf16)b.z; v[7]=(__bf16)b.w;
        *(bf16x8*)&dst[i] = v;
    }
}

// ---------------------------------------------------------------------------
// Kernel 1: router — logits, top-2, softmax(top-2), compact per-expert lists
// ---------------------------------------------------------------------------
__global__ __launch_bounds__(256)
void moe_router(const float* __restrict__ tokens,
                const float* __restrict__ router_w,
                int*   __restrict__ counts,
                int*   __restrict__ tok_list,
                float* __restrict__ prob_list) {
    __shared__ float rw[NEXP * HDIM];  // 32 KB
    const int tid = threadIdx.x;
    for (int i = tid * 4; i < NEXP * HDIM; i += 256 * 4) {
        *(float4*)&rw[i] = *(const float4*)&router_w[i];
    }
    __syncthreads();

    const int wave = tid >> 5;
    const int lane = tid & 31;
    const int token = blockIdx.x * 8 + wave;

    float acc[NEXP];
#pragma unroll
    for (int e = 0; e < NEXP; ++e) acc[e] = 0.0f;

    const float* trow = tokens + (size_t)token * HDIM;
    for (int h = lane * 4; h < HDIM; h += 32 * 4) {
        float4 tv = *(const float4*)&trow[h];
#pragma unroll
        for (int e = 0; e < NEXP; ++e) {
            float4 wv = *(const float4*)&rw[e * HDIM + h];
            acc[e] += tv.x * wv.x + tv.y * wv.y + tv.z * wv.z + tv.w * wv.w;
        }
    }
#pragma unroll
    for (int e = 0; e < NEXP; ++e) {
#pragma unroll
        for (int off = 16; off > 0; off >>= 1)
            acc[e] += __shfl_xor(acc[e], off, 32);
    }

    if (lane == 0) {
        int i0 = 0; float s0 = acc[0];
#pragma unroll
        for (int e = 1; e < NEXP; ++e) if (acc[e] > s0) { s0 = acc[e]; i0 = e; }
        int i1 = (i0 == 0) ? 1 : 0; float s1 = acc[i1];
#pragma unroll
        for (int e = 0; e < NEXP; ++e)
            if (e != i0 && acc[e] > s1) { s1 = acc[e]; i1 = e; }
        float p0 = 1.0f / (1.0f + expf(s1 - s0));   // softmax over top-2
        float p1 = 1.0f - p0;
        int pos0 = atomicAdd(&counts[i0], 1);
        tok_list[i0 * N_TOK + pos0]  = token;
        prob_list[i0 * N_TOK + pos0] = p0;
        int pos1 = atomicAdd(&counts[i1], 1);
        tok_list[i1 * N_TOK + pos1]  = token;
        prob_list[i1 * N_TOK + pos1] = p1;
    }
}

// ---------------------------------------------------------------------------
// B-fragment loader: lane holds 16 consecutive K-values.
// ---------------------------------------------------------------------------
template <typename WT>
__device__ inline bf16x16 load_b_frag(const WT* wrow) {
    if constexpr (sizeof(WT) == 2) {
        return *(const bf16x16*)wrow;          // one aligned 32B vector load
    } else {
        float4 f0 = *(const float4*)(wrow + 0);
        float4 f1 = *(const float4*)(wrow + 4);
        float4 f2 = *(const float4*)(wrow + 8);
        float4 f3 = *(const float4*)(wrow + 12);
        bf16x16 b;
        b[0]=(__bf16)f0.x;  b[1]=(__bf16)f0.y;  b[2]=(__bf16)f0.z;  b[3]=(__bf16)f0.w;
        b[4]=(__bf16)f1.x;  b[5]=(__bf16)f1.y;  b[6]=(__bf16)f1.z;  b[7]=(__bf16)f1.w;
        b[8]=(__bf16)f2.x;  b[9]=(__bf16)f2.y;  b[10]=(__bf16)f2.z; b[11]=(__bf16)f2.w;
        b[12]=(__bf16)f3.x; b[13]=(__bf16)f3.y; b[14]=(__bf16)f3.z; b[15]=(__bf16)f3.w;
        return b;
    }
}

// ---------------------------------------------------------------------------
// Kernel 2: per-expert FFN on 64-token tiles, bf16 WMMA with f32 accumulate.
// grid.x = NEXP * MAXT ; block = 512 (16 column-waves; 4 row-groups per wave,
// so each loaded B-fragment feeds 4 WMMAs -> 4x less L2 weight traffic).
// ---------------------------------------------------------------------------
template <typename WT>
__global__ __launch_bounds__(BLOCK)
void moe_expert_ffn(const float* __restrict__ tokens,
                    const WT*    __restrict__ w1,
                    const float* __restrict__ b1,
                    const WT*    __restrict__ w2,
                    const float* __restrict__ b2,
                    const int*   __restrict__ counts,
                    const int*   __restrict__ tok_list,
                    const float* __restrict__ prob_list,
                    float* __restrict__ out) {
    __shared__ __bf16 Xs[TM * XS_STRIDE];   // ~132 KB  token tile (bf16)
    __shared__ __bf16 Hs[TM * HS_STRIDE];   // ~67 KB   fc1 activation chunk (bf16)
    __shared__ int   sTok[TM];
    __shared__ float sProb[TM];

    const int e = blockIdx.x >> 7;          // / MAXT (=128)
    const int t = blockIdx.x & (MAXT - 1);
    const int cnt_all = counts[e];
    if (t * TM >= cnt_all) return;
    const int base = t * TM;
    const int cnt  = min(TM, cnt_all - base);

    const int tid  = threadIdx.x;
    const int cw   = tid >> 5;              // column-wave index 0..15
    const int lane = tid & 31;
    const int col  = lane & 15;             // A-row / B-col index within 16x16 tile
    const int hi   = lane >> 4;             // K-half selector (WMMA 16-bit layout)

    if (tid < TM) {
        int tk = 0; float p = 0.0f;
        if (tid < cnt) {
            tk = tok_list[e * N_TOK + base + tid];
            p  = prob_list[e * N_TOK + base + tid];
        }
        sTok[tid] = tk; sProb[tid] = p;     // padded rows: token 0, weight 0
    }
    __syncthreads();

    // ---- gather token rows -> LDS bf16 (wave cw handles rows cw+16*rr) ----
#pragma unroll
    for (int rr = 0; rr < 4; ++rr) {
        int r = cw + rr * 16;
        const float* src = tokens + (size_t)sTok[r] * HDIM;
        __bf16* dst = &Xs[r * XS_STRIDE];
        for (int c = lane * 4; c < HDIM; c += 128) {
            float4 v = *(const float4*)&src[c];
            dst[c + 0] = (__bf16)v.x; dst[c + 1] = (__bf16)v.y;
            dst[c + 2] = (__bf16)v.z; dst[c + 3] = (__bf16)v.w;
        }
    }
    __syncthreads();

    // A fragment from LDS: lane half hi holds 16 contiguous K (k = 16*hi + s), matching B.
    auto load_a = [&](const __bf16* lds, int stride, int rowbase, int kt) -> bf16x16 {
        const __bf16* p = lds + (rowbase + col) * stride + kt + hi * 16;
        bf16x8 lo = *(const bf16x8*)p;
        bf16x8 up = *(const bf16x8*)(p + 8);
        bf16x16 a;
#pragma unroll
        for (int i = 0; i < 8; ++i) { a[i] = lo[i]; a[8 + i] = up[i]; }
        return a;
    };

    const WT*    w1e = w1 + (size_t)e * H2 * HDIM;
    const WT*    w2e = w2 + (size_t)e * HDIM * H2;
    const float* b1e = b1 + (size_t)e * H2;
    const float* b2e = b2 + (size_t)e * HDIM;

    // fc2 weight row base pointers for this wave's 4 output col-tiles
    const WT* w2row[4];
#pragma unroll
    for (int jj = 0; jj < 4; ++jj)
        w2row[jj] = w2e + (size_t)((cw + 16 * jj) * 16 + col) * H2 + hi * 16;

    f32x8 zero8 = {0.f,0.f,0.f,0.f,0.f,0.f,0.f,0.f};
    f32x8 yacc[4][4];                        // [col-tile jj][row-group r]
#pragma unroll
    for (int jj = 0; jj < 4; ++jj)
#pragma unroll
        for (int r = 0; r < 4; ++r) yacc[jj][r] = zero8;

    for (int chunk = 0; chunk < H2 / HCHUNK; ++chunk) {      // 4 chunks of 512
        // ---- fc1: 32 col-tiles/chunk; this wave owns 2 (all 4 row-groups) -
#pragma unroll
        for (int pass = 0; pass < 2; ++pass) {
            int jt   = cw + 16 * pass;                       // chunk-local tile
            int lcol = jt * 16 + col;
            const WT* wrow = w1e + (size_t)(chunk * HCHUNK + lcol) * HDIM + hi * 16;
            f32x8 c4[4];
#pragma unroll
            for (int r = 0; r < 4; ++r) c4[r] = zero8;
#pragma unroll 2
            for (int kt = 0; kt < HDIM; kt += 32) {
                __builtin_prefetch(wrow + kt + 256, 0, 1);   // global_prefetch_b8
                bf16x16 b = load_b_frag<WT>(wrow + kt);      // shared by 4 WMMAs
#pragma unroll
                for (int r = 0; r < 4; ++r) {
                    bf16x16 a = load_a(Xs, XS_STRIDE, r * 16, kt);
                    c4[r] = __builtin_amdgcn_wmma_f32_16x16x32_bf16(
                                false, a, false, b, (short)0, c4[r], false, false);
                }
            }
            float bias = b1e[chunk * HCHUNK + lcol];
#pragma unroll
            for (int r = 0; r < 4; ++r) {                    // exact GELU -> Hs
#pragma unroll
                for (int i = 0; i < 8; ++i) {
                    int m = r * 16 + i + hi * 8;
                    float x = c4[r][i] + bias;
                    float g = 0.5f * x * (1.0f + erff(x * 0.70710678118654752f));
                    Hs[m * HS_STRIDE + lcol] = (__bf16)g;
                }
            }
        }
        __syncthreads();
        // ---- fc2 partial: 4 col-tiles x 4 row-groups; A loaded once per kt
#pragma unroll 2
        for (int kt = 0; kt < HCHUNK; kt += 32) {
            bf16x16 a4[4];
#pragma unroll
            for (int r = 0; r < 4; ++r) a4[r] = load_a(Hs, HS_STRIDE, r * 16, kt);
#pragma unroll
            for (int jj = 0; jj < 4; ++jj) {
                const WT* wp = w2row[jj] + chunk * HCHUNK + kt;
                __builtin_prefetch(wp + 256, 0, 1);
                bf16x16 b = load_b_frag<WT>(wp);             // shared by 4 WMMAs
#pragma unroll
                for (int r = 0; r < 4; ++r)
                    yacc[jj][r] = __builtin_amdgcn_wmma_f32_16x16x32_bf16(
                                      false, a4[r], false, b, (short)0,
                                      yacc[jj][r], false, false);
            }
        }
        __syncthreads();                       // Hs reused next chunk
    }

    // -------- epilogue: bias, top-k prob scale, scatter-add ---------------
#pragma unroll
    for (int jj = 0; jj < 4; ++jj) {
        int ocol = (cw + 16 * jj) * 16 + col;
        float bias = b2e[ocol];
#pragma unroll
        for (int r = 0; r < 4; ++r) {
#pragma unroll
            for (int i = 0; i < 8; ++i) {
                int m = r * 16 + i + hi * 8;
                if (m < cnt) {
                    float v = sProb[m] * (yacc[jj][r][i] + bias);
                    atomicAdd(&out[(size_t)sTok[m] * HDIM + ocol], v);
                }
            }
        }
    }
}

// ---------------------------------------------------------------------------
// Host entry
// ---------------------------------------------------------------------------
extern "C" void kernel_launch(void* const* d_in, const int* in_sizes, int n_in,
                              void* d_out, int out_size, void* d_ws, size_t ws_size,
                              hipStream_t stream) {
    const float* tokens   = (const float*)d_in[0];
    const float* router_w = (const float*)d_in[1];
    const float* w1       = (const float*)d_in[2];
    const float* b1       = (const float*)d_in[3];
    const float* w2       = (const float*)d_in[4];
    const float* b2       = (const float*)d_in[5];
    float* out = (float*)d_out;

    // workspace layout:
    // [counts 8i][pad->256B][tok_list E*N i32][prob_list E*N f32][w1 bf16][w2 bf16]
    const size_t listBytes = (size_t)NEXP * N_TOK * 4;
    const size_t wElems    = (size_t)NEXP * H2 * HDIM;       // per weight tensor
    int*    counts    = (int*)d_ws;
    int*    tok_list  = (int*)((char*)d_ws + 256);
    float*  prob_list = (float*)((char*)d_ws + 256 + listBytes);
    __bf16* w1b       = (__bf16*)((char*)d_ws + 256 + 2 * listBytes);
    __bf16* w2b       = w1b + wElems;
    const size_t need = 256 + 2 * listBytes + 2 * wElems * sizeof(__bf16);

    moe_zero<<<(N_TOK * HDIM + 255) / 256, 256, 0, stream>>>(out, counts, N_TOK * HDIM);
    moe_router<<<N_TOK / 8, 256, 0, stream>>>(tokens, router_w, counts, tok_list, prob_list);

    if (ws_size >= need) {
        // stage weights as bf16 (one HBM pass) then run the fast path
        int nconv = (int)wElems;
        moe_cvt_bf16<<<(nconv / 8 + 255) / 256, 256, 0, stream>>>(w1, w1b, nconv);
        moe_cvt_bf16<<<(nconv / 8 + 255) / 256, 256, 0, stream>>>(w2, w2b, nconv);
        moe_expert_ffn<__bf16><<<NEXP * MAXT, BLOCK, 0, stream>>>(
            tokens, w1b, b1, w2b, b2, counts, tok_list, prob_list, out);
    } else {
        // fallback: stream fp32 weights with in-register conversion
        moe_expert_ffn<float><<<NEXP * MAXT, BLOCK, 0, stream>>>(
            tokens, w1, b1, w2, b2, counts, tok_list, prob_list, out);
    }
}
